// Separator_28424093565064
// MI455X (gfx1250) — matrix operand order repaired and checked by reading, not compiled, not run on hardware
//
#include <hip/hip_runtime.h>
#include <hip/hip_bf16.h>
#include <stdint.h>

// ---------------------------------------------------------------------------
// Separator (open-unmix style): STFT -> |.| -> per-target GEMM(+bias,ReLU)
// -> multiply by phase -> ISTFT.  Dominant work: 58-GFLOP GEMM done with
// v_wmma_f32_16x16x32_bf16 on gfx1250 (wave32).  GEMM stages fragments
// through LDS with global_load_async_to_lds_b128 (ASYNCcnt double-buffer),
// 8 waves share a 128x64 C block -> 43 FLOP/byte of global traffic.
// All intermediates (~175 MB) are L2-resident on MI455X (192 MB L2).
// ---------------------------------------------------------------------------

#define NFFT      4096
#define HOP       1024
#define NBINS     2049
#define NFRAMES   431
#define TAUDIO    441000
#define PADQ      2048                         // NFFT/2
#define OUTLEN    (NFFT + HOP * (NFRAMES - 1)) // 444416
#define TWO_PI    6.28318530717958647692f
#define PI_F      3.14159265358979323846f

// GEMM tiling: M = 2049 (g) -> 136 padded tiles of 16 (17 blocks x 8 tiles),
// N = 431 (t) -> 28 padded tiles of 16 (7 blocks x 4 tiles),
// K = 2049 (f) -> 65 chunks of 32 (bf16 WMMA K=32).
#define MTP      136
#define NTP      28
#define KT       65
#define MBLOCKS  17
#define NBLOCKS  7
#define FRAG_B   1024                 // bytes per packed fragment (32 lanes x 32B)
#define TILE_LDS (12 * FRAG_B)        // 8 A-frags + 4 B-frags per K-step

typedef __attribute__((ext_vector_type(16))) __bf16 v16bf;
typedef __attribute__((ext_vector_type(8)))  float  v8f;
typedef __attribute__((ext_vector_type(4)))  int    v4i;

typedef __attribute__((address_space(1))) v4i gas_v4i;   // global int4
typedef __attribute__((address_space(3))) v4i las_v4i;   // LDS int4

__device__ __forceinline__ unsigned short f32_to_bf16_rne(float f) {
    union { float f; uint32_t u; } v; v.f = f;
    uint32_t u = v.u;
    uint32_t r = u + 0x7FFFu + ((u >> 16) & 1u);   // round-to-nearest-even
    return (unsigned short)(r >> 16);
}

__device__ __forceinline__ float hann_w(int k) {
    return 0.5f - 0.5f * __cosf(TWO_PI * (float)k / (float)NFFT);
}

// In-place radix-2 DIT FFT over 4096 complex values in LDS; input must be
// pre-scattered in bit-reversed order.  256 threads, forward (e^{-i...}).
__device__ void fft4096(float* sre, float* sim) {
    const int tid = threadIdx.x;
    for (int stage = 0; stage < 12; ++stage) {
        const int mh = 1 << stage;
        for (int bf = tid; bf < 2048; bf += 256) {
            const int pos = bf & (mh - 1);
            const int i0  = ((bf >> stage) << (stage + 1)) + pos;
            const int i1  = i0 + mh;
            float s, c;
            __sincosf(-PI_F * (float)pos / (float)mh, &s, &c);
            const float xr = sre[i1], xi = sim[i1];
            const float tr = xr * c - xi * s;
            const float ti = xr * s + xi * c;
            const float ur = sre[i0], ui = sim[i0];
            sre[i1] = ur - tr; sim[i1] = ui - ti;
            sre[i0] = ur + tr; sim[i0] = ui + ti;
        }
        __syncthreads();
    }
}

// --------------------------- STFT ------------------------------------------
__global__ __launch_bounds__(256) void stft_kernel(
    const float* __restrict__ audio,
    float* __restrict__ ph_re, float* __restrict__ ph_im,
    float* __restrict__ xmag)
{
    __shared__ float sre[NFFT];
    __shared__ float sim[NFFT];
    const int t  = blockIdx.x % NFRAMES;
    const int nc = blockIdx.x / NFRAMES;
    const float* __restrict__ a = audio + (size_t)nc * TAUDIO;
    const int tid = threadIdx.x;

    for (int k = tid; k < NFFT; k += 256) {          // reflect pad + window
        int s = t * HOP + k - PADQ;
        if (s < 0) s = -s;
        if (s >= TAUDIO) s = 2 * TAUDIO - 2 - s;
        const int r = __brev(k) >> 20;               // 12-bit bit reversal
        sre[r] = a[s] * hann_w(k);
        sim[r] = 0.0f;
    }
    __syncthreads();
    fft4096(sre, sim);

    const size_t base = (size_t)nc * NBINS * NFRAMES + t;
    for (int f = tid; f < NBINS; f += 256) {
        const float re = sre[f], im = sim[f];
        const float mag = sqrtf(re * re + im * im);
        const float inv = 1.0f / (mag + 1e-10f);
        const size_t o = base + (size_t)f * NFRAMES;
        xmag[o]  = mag;
        ph_re[o] = re * inv;
        ph_im[o] = im * inv;
    }
}

// --------------------------- Fragment packing ------------------------------
// A-matrix (16x32 bf16) per-lane layout (ISA 7.12.2): lanes 0-15 hold M=lane,
// dwords 0-3 -> K 0..7, dwords 4-7 -> K 16..23; lanes 16-31 shift K by +8.
__global__ __launch_bounds__(256) void pack_w_kernel(
    const float* __restrict__ W, uint32_t* __restrict__ wfrag)
{
    const int gid = blockIdx.x * 256 + threadIdx.x;  // frag*32 + lane
    const int NFR = 4 * MTP * KT;
    if (gid >= NFR * 32) return;
    const int lane = gid & 31;
    const int frag = gid >> 5;
    const int kt = frag % KT;
    const int mt = (frag / KT) % MTP;
    const int j  = frag / (KT * MTP);
    const int m  = mt * 16 + (lane & 15);
    const int khalf = (lane >= 16) ? 8 : 0;
    const bool mv = (m < NBINS);
    const float* __restrict__ wrow = W + ((size_t)j * NBINS + m) * NBINS;
    uint32_t out[8];
#pragma unroll
    for (int d = 0; d < 8; ++d) {
        uint32_t pk = 0;
#pragma unroll
        for (int h = 0; h < 2; ++h) {
            const int i = d * 2 + h;                            // 0..15
            const int k = kt * 32 + ((i < 8) ? (khalf + i) : (16 + khalf + i - 8));
            const float v = (mv && k < NBINS) ? wrow[k] : 0.0f;
            pk |= (uint32_t)f32_to_bf16_rne(v) << (16 * h);
        }
        out[d] = pk;
    }
    uint32_t* dst = wfrag + (size_t)gid * 8;
    ((uint4*)dst)[0] = make_uint4(out[0], out[1], out[2], out[3]);
    ((uint4*)dst)[1] = make_uint4(out[4], out[5], out[6], out[7]);
}

// B-matrix (32x16 bf16): lanes 0-15 hold N=lane with K 0..7,16..23 (2 per dword),
// lanes 16-31 hold N=lane-16 with K 8..15,24..31.
__global__ __launch_bounds__(256) void pack_x_kernel(
    const float* __restrict__ xmag, uint32_t* __restrict__ xfrag)
{
    const int gid = blockIdx.x * 256 + threadIdx.x;
    const int NFR = 4 * KT * NTP;
    if (gid >= NFR * 32) return;
    const int lane = gid & 31;
    const int frag = gid >> 5;
    const int nt = frag % NTP;
    const int kt = (frag / NTP) % KT;
    const int nc = frag / (NTP * KT);
    const int tcol = nt * 16 + (lane & 15);
    const int kbase = kt * 32 + ((lane >= 16) ? 8 : 0);
    const bool tv = (tcol < NFRAMES);
    const float* __restrict__ xp = xmag + (size_t)nc * NBINS * NFRAMES + tcol;
    uint32_t out[8];
#pragma unroll
    for (int d = 0; d < 8; ++d) {
        uint32_t pk = 0;
#pragma unroll
        for (int h = 0; h < 2; ++h) {
            const int k = kbase + ((d * 2 + h < 8) ? (d * 2 + h) : (8 + d * 2 + h));
            const float v = (tv && k < NBINS) ? xp[(size_t)k * NFRAMES] : 0.0f;
            pk |= (uint32_t)f32_to_bf16_rne(v) << (16 * h);
        }
        out[d] = pk;
    }
    uint32_t* dst = xfrag + (size_t)gid * 8;
    ((uint4*)dst)[0] = make_uint4(out[0], out[1], out[2], out[3]);
    ((uint4*)dst)[1] = make_uint4(out[4], out[5], out[6], out[7]);
}

// --------------------------- WMMA GEMM with async-LDS staging --------------
// Block = 8 waves (4M x 2N) computing a 128x64 C block; per K-step the block
// stages 8 A-frags + 4 B-frags (12 KB) into LDS via
// global_load_async_to_lds_b128 (3 x 512B copies per wave), double-buffered
// on ASYNCcnt; each wave then does 4 WMMAs on ds_load'ed fragments.
__global__ __launch_bounds__(256) void gemm_kernel(
    const uint32_t* __restrict__ wfrag, const uint32_t* __restrict__ xfrag,
    const float* __restrict__ bias, float* __restrict__ V)
{
    __shared__ __align__(32) char smem[2 * TILE_LDS];   // 24 KB double buffer

    const int tid  = threadIdx.x;
    const int lane = tid & 31;
    const int wave = tid >> 5;                   // 0..7
    const int wm = wave >> 1, wn = wave & 1;     // 4 x 2 wave grid

    const int bx = blockIdx.x;
    const int nb = bx % NBLOCKS;
    const int mb = (bx / NBLOCKS) % MBLOCKS;
    const int nc = (bx / (NBLOCKS * MBLOCKS)) & 3;
    const int j  =  bx / (NBLOCKS * MBLOCKS * 4);
    const int mtb = mb * 8;                      // first of 8 M tiles
    const int ntb = nb * 4;                      // first of 4 N tiles

    const char* __restrict__ wj =
        (const char*)wfrag + (size_t)j * MTP * KT * FRAG_B;
    const char* __restrict__ xn =
        (const char*)xfrag + (size_t)nc * KT * NTP * FRAG_B;

    // Issue this wave's 3 async 512-B copies for K-step kt into buffer par.
    auto issue = [&](int par, int kt) {
#pragma unroll
        for (int s = 0; s < 3; ++s) {
            const int li   = wave * 3 + s;       // 0..23
            const int frag = li >> 1;            // 0..11
            const int off  = (li & 1) * 512 + lane * 16;
            const char* g = (frag < 8)
                ? wj + ((size_t)(mtb + frag) * KT + kt) * FRAG_B + off
                : xn + ((size_t)kt * NTP + (ntb + frag - 8)) * FRAG_B + off;
            char* l = smem + par * TILE_LDS + frag * FRAG_B + off;
            __builtin_amdgcn_global_load_async_to_lds_b128(
                (gas_v4i*)(uintptr_t)g,
                (las_v4i*)(uint32_t)(uintptr_t)l, 0, 0);
        }
    };

    v8f acc00 = {0.f,0.f,0.f,0.f,0.f,0.f,0.f,0.f};
    v8f acc01 = acc00, acc10 = acc00, acc11 = acc00;

    const int a0i = wm * 2, a1i = wm * 2 + 1;          // LDS fragment slots
    const int b0i = 8 + wn * 2, b1i = 9 + wn * 2;

    auto compute = [&](int par) {
        const char* buf = smem + par * TILE_LDS;
        const v16bf a0 = *(const v16bf*)(buf + a0i * FRAG_B + lane * 32);
        const v16bf a1 = *(const v16bf*)(buf + a1i * FRAG_B + lane * 32);
        const v16bf b0 = *(const v16bf*)(buf + b0i * FRAG_B + lane * 32);
        const v16bf b1 = *(const v16bf*)(buf + b1i * FRAG_B + lane * 32);
        acc00 = __builtin_amdgcn_wmma_f32_16x16x32_bf16(
            false, a0, false, b0, (short)0, acc00, false, false);
        acc01 = __builtin_amdgcn_wmma_f32_16x16x32_bf16(
            false, a0, false, b1, (short)0, acc01, false, false);
        acc10 = __builtin_amdgcn_wmma_f32_16x16x32_bf16(
            false, a1, false, b0, (short)0, acc10, false, false);
        acc11 = __builtin_amdgcn_wmma_f32_16x16x32_bf16(
            false, a1, false, b1, (short)0, acc11, false, false);
    };

    issue(0, 0);                                  // prologue
#pragma unroll 1
    for (int kt = 0; kt < KT - 1; ++kt) {
        issue((kt + 1) & 1, kt + 1);              // prefetch next K-step
        __builtin_amdgcn_s_wait_asynccnt(3);      // this wave's current-buf copies done
        __syncthreads();                          // ... all waves' copies done
        compute(kt & 1);
        __syncthreads();                          // reads done before buffer reuse
    }
    __builtin_amdgcn_s_wait_asynccnt(0);          // last K-step
    __syncthreads();
    compute((KT - 1) & 1);

    // C/D layout: VGPR r, lanes 0-15 -> M=r, N=lane; lanes 16-31 -> M=r+8.
    const int njc = ((nc >> 1) * 8) + j * 2 + (nc & 1);       // n*8 + j*2 + c
    const int mt0 = mtb + wm * 2, nt0 = ntb + wn * 2;
    const int mofs = (lane >= 16) ? 8 : 0;
    const int nofs = lane & 15;
    const float* __restrict__ brow = bias + j * NBINS;
    float* __restrict__ vb = V + (size_t)njc * NBINS * NFRAMES;

#pragma unroll
    for (int q = 0; q < 4; ++q) {
        const v8f acc = (q == 0) ? acc00 : (q == 1) ? acc01
                      : (q == 2) ? acc10 : acc11;
        const int g_lo = (mt0 + (q >> 1)) * 16 + mofs;
        const int tt   = (nt0 + (q & 1)) * 16 + nofs;
        if (tt < NFRAMES) {
#pragma unroll
            for (int r = 0; r < 8; ++r) {
                const int g = g_lo + r;
                if (g < NBINS)
                    vb[(size_t)g * NFRAMES + tt] = fmaxf(acc[r] + brow[g], 0.0f);
            }
        }
    }
}

// --------------------------- ISTFT + overlap-add ---------------------------
__global__ __launch_bounds__(256) void istft_kernel(
    const float* __restrict__ V, const float* __restrict__ ph_re,
    const float* __restrict__ ph_im, float* __restrict__ acc)
{
    __shared__ float sre[NFFT];
    __shared__ float sim[NFFT];
    const int t   = blockIdx.x % NFRAMES;
    const int njc = blockIdx.x / NFRAMES;            // n*8 + j*2 + c
    const int n = njc >> 3, c = njc & 1;
    const int nc = n * 2 + c;
    const int tid = threadIdx.x;
    const size_t vb = (size_t)njc * NBINS * NFRAMES + t;
    const size_t pb = (size_t)nc  * NBINS * NFRAMES + t;

    // Build conj(Hermitian spectrum) bit-reversed: ifft = conj(fft(conj(X)))/N
    for (int f = tid; f < NBINS; f += 256) {
        const float v  = V[vb + (size_t)f * NFRAMES];
        const float pr = ph_re[pb + (size_t)f * NFRAMES];
        const float pi = ph_im[pb + (size_t)f * NFRAMES];
        const float xr = v * pr, xi = v * pi;
        const int r = __brev(f) >> 20;
        sre[r] = xr; sim[r] = -xi;
        if (f >= 1 && f < NFFT / 2) {
            const int r2 = __brev(NFFT - f) >> 20;
            sre[r2] = xr; sim[r2] = xi;
        }
    }
    __syncthreads();
    fft4096(sre, sim);

    const float scale = 1.0f / (float)NFFT;
    float* __restrict__ dst = acc + (size_t)njc * OUTLEN + (size_t)t * HOP;
    for (int k = tid; k < NFFT; k += 256)
        atomicAdd(&dst[k], sre[k] * scale * hann_w(k));
}

// --------------------------- window-sum normalize + crop -------------------
__global__ __launch_bounds__(256) void normalize_kernel(
    const float* __restrict__ acc, float* __restrict__ out)
{
    const int gid = blockIdx.x * 256 + threadIdx.x;
    if (gid >= 16 * TAUDIO) return;
    const int i   = gid % TAUDIO;
    const int njc = gid / TAUDIO;
    const int p = i + PADQ;
    const float y = acc[(size_t)njc * OUTLEN + p];
    int tmax = p >> 10; if (tmax > NFRAMES - 1) tmax = NFRAMES - 1;
    int tmin = (p >= NFFT) ? ((p - (NFFT - 1) + HOP - 1) >> 10) : 0;
    float wsq = 0.0f;
    for (int tf = tmin; tf <= tmax; ++tf) {
        const float w = hann_w(p - tf * HOP);
        wsq += w * w;
    }
    out[gid] = y / (wsq > 1e-11f ? wsq : 1.0f);
}

// ---------------------------------------------------------------------------
extern "C" void kernel_launch(void* const* d_in, const int* in_sizes, int n_in,
                              void* d_out, int out_size, void* d_ws, size_t ws_size,
                              hipStream_t stream) {
    const float* audio = (const float*)d_in[0];     // (2,2,441000)
    const float* W     = (const float*)d_in[1];     // (4,2049,2049)
    const float* bias  = (const float*)d_in[2];     // (4,2049)
    float* out = (float*)d_out;                     // (2,4,2,441000)

    // Workspace carve-out (256B aligned slices).
    char* base = (char*)d_ws;
    size_t off = 0;
    auto take = [&](size_t bytes) -> char* {
        char* p = base + off;
        off = (off + bytes + 255) & ~(size_t)255;
        return p;
    };
    const size_t SP = (size_t)NBINS * NFRAMES;      // 883,119 per (n,c)
    float*    ph_re = (float*)take(4 * SP * sizeof(float));
    float*    ph_im = (float*)take(4 * SP * sizeof(float));
    float*    xmag  = (float*)take(4 * SP * sizeof(float));
    uint32_t* wfrag = (uint32_t*)take((size_t)4 * MTP * KT * FRAG_B);
    uint32_t* xfrag = (uint32_t*)take((size_t)4 * KT * NTP * FRAG_B);
    float*    V     = (float*)take(16 * SP * sizeof(float));
    float*    acc   = (float*)take((size_t)16 * OUTLEN * sizeof(float));
    (void)ws_size; (void)n_in; (void)in_sizes; (void)out_size;

    // Zero only the overlap-add accumulator (deterministic per call).
    (void)hipMemsetAsync(acc, 0, (size_t)16 * OUTLEN * sizeof(float), stream);

    stft_kernel<<<4 * NFRAMES, 256, 0, stream>>>(audio, ph_re, ph_im, xmag);

    {   // pack W -> bf16 A fragments (padded to 136 M-tiles)
        const int threads = 4 * MTP * KT * 32;
        pack_w_kernel<<<(threads + 255) / 256, 256, 0, stream>>>(W, wfrag);
    }
    {   // pack |X| -> bf16 B fragments (padded to 28 N-tiles)
        const int threads = 4 * KT * NTP * 32;
        pack_x_kernel<<<(threads + 255) / 256, 256, 0, stream>>>(xmag, xfrag);
    }

    // 4(j) * 4(nc) * 17(mb) * 7(nb) = 1,904 blocks of 8 waves.
    gemm_kernel<<<4 * 4 * MBLOCKS * NBLOCKS, 256, 0, stream>>>(wfrag, xfrag, bias, V);

    istft_kernel<<<16 * NFRAMES, 256, 0, stream>>>(V, ph_re, ph_im, acc);

    normalize_kernel<<<(16 * TAUDIO + 255) / 256, 256, 0, stream>>>(acc, out);
}
